// ChannelAttentionV2_89687507075829
// MI455X (gfx1250) — matrix operand order repaired and checked by reading, not compiled, not run on hardware
//
#include <hip/hip_runtime.h>
#include <hip/hip_bf16.h>
#include <stdint.h>

// ---------------------------------------------------------------------------
// Channel attention (B=16, N=4096, C=768, H=8, D=96) for gfx1250 (MI455X).
// GEMMs via v_wmma_f32_16x16x32_bf16; big projections stage A tiles in LDS
// with global_load_async_to_lds_b128 (ASYNCcnt), double-buffered with a
// 2x-unrolled k-loop (no register rotation, no WMMA-hazard NOPs).
// ---------------------------------------------------------------------------

#define BATCH   16
#define NTOK    4096
#define CDIM    768
#define NHEAD   8
#define HDIM    96
#define TROWS   (BATCH * NTOK)   // 65536
#define C3      (3 * CDIM)       // 2304
#define LDS_STRIDE 40            // padded row stride (shorts) for 32-col tiles

typedef __attribute__((ext_vector_type(16))) __bf16   bf16x16;
typedef __attribute__((ext_vector_type(8)))  float    f32x8;
typedef __attribute__((ext_vector_type(4)))  unsigned int u32x4;

__device__ __forceinline__ unsigned short f32_to_bf16(float f) {
    union { float f; unsigned u; } v; v.f = f;
    unsigned r = v.u + 0x7FFFu + ((v.u >> 16) & 1u);   // round-to-nearest-even
    return (unsigned short)(r >> 16);
}

// A-fragment: 16x32 bf16 tile, row-major source, tile origin pre-applied.
// Lane l (<16): M=l, K={0..7,16..23}; lane l+16: M=l, K={8..15,24..31}.
__device__ __forceinline__ bf16x16 load_a_frag(const unsigned short* __restrict__ A,
                                               int lda, int lane) {
    int hl = lane >> 4, m = lane & 15;
    const unsigned short* p = A + (size_t)m * lda + hl * 8;
    union { bf16x16 v; u32x4 q[2]; } f;
    f.q[0] = *(const u32x4*)(p);        // K = 8*hl .. 8*hl+7
    f.q[1] = *(const u32x4*)(p + 16);   // K = 16+8*hl .. 16+8*hl+7
    return f.v;
}

// B-fragment: 32x16 bf16 tile read from K-contiguous B^T (ldb = K-stride).
// Lane l (<16): N=l, K={0..7,16..23}; lane l+16: N=l, K={8..15,24..31}.
__device__ __forceinline__ bf16x16 load_b_frag(const unsigned short* __restrict__ Bt,
                                               int ldb, int lane) {
    int hl = lane >> 4, n = lane & 15;
    const unsigned short* p = Bt + (size_t)n * ldb + hl * 8;
    union { bf16x16 v; u32x4 q[2]; } f;
    f.q[0] = *(const u32x4*)(p);
    f.q[1] = *(const u32x4*)(p + 16);
    return f.v;
}

// A-fragment from an LDS tile with row stride LDS_STRIDE (shorts).
__device__ __forceinline__ bf16x16 load_a_frag_lds(const unsigned short* tile,
                                                   int lane) {
    int hl = lane >> 4, m = lane & 15;
    const unsigned short* p = tile + m * LDS_STRIDE + hl * 8;
    union { bf16x16 v; u32x4 q[2]; } f;
    f.q[0] = *(const u32x4*)(p);
    f.q[1] = *(const u32x4*)(p + 16);
    return f.v;
}

__device__ __forceinline__ f32x8 wmma_bf16(bf16x16 a, bf16x16 b, f32x8 c) {
    return __builtin_amdgcn_wmma_f32_16x16x32_bf16(false, a, false, b,
                                                   (short)0, c, false, false);
}

// Async copy of one 16-byte chunk from global into LDS (ASYNCcnt-tracked).
__device__ __forceinline__ void async_stage16(const unsigned short* g,
                                              unsigned short* l) {
    unsigned loff = (unsigned)(uintptr_t)l;          // low 32 bits = LDS offset
    asm volatile("global_load_async_to_lds_b128 %0, %1, off"
                 :: "v"(loff), "v"((const void*)g) : "memory");
}
__device__ __forceinline__ void wait_async0() {
    asm volatile("s_wait_asynccnt 0" ::: "memory");
}

// ---------------------------------------------------------------------------
// f32 -> bf16 conversion (grid-stride)
// ---------------------------------------------------------------------------
__global__ __launch_bounds__(256) void cvt_bf16_kernel(const float* __restrict__ src,
                                                       unsigned short* __restrict__ dst,
                                                       int n) {
    int i = blockIdx.x * blockDim.x + threadIdx.x;
    int stride = gridDim.x * blockDim.x;
    for (; i < n; i += stride) dst[i] = f32_to_bf16(src[i]);
}

// ---------------------------------------------------------------------------
// QKV projection: [65536 x 768] @ [768 x 2304] + bias.
// Block = 8 waves sharing a 64-row A strip (block tile 64x256); A k-tiles
// staged to LDS asynchronously. k-loop unrolled 2x: even step consumes
// buffer0/b0 while staging kk+1 into buffer1/b1 and vice versa (no reg movs).
// Scatters q (scaled, [B,H,D,N]), k ([B,H,D,N]), v ([B,H,N,D]) as bf16.
// ---------------------------------------------------------------------------
__global__ __launch_bounds__(256) void qkv_gemm_kernel(
        const unsigned short* __restrict__ xb,
        const unsigned short* __restrict__ wq,     // [2304 x 768] (== B^T)
        const float* __restrict__ bias,            // [2304]
        unsigned short* __restrict__ q_t,
        unsigned short* __restrict__ k_t,
        unsigned short* __restrict__ v_b) {
    __shared__ __align__(16) unsigned short smem[2][64 * LDS_STRIDE];
    const int tid  = threadIdx.x;
    const int lane = tid & 31, wv = tid >> 5;
    const int bm = blockIdx.x / (C3 / 256);        // 1024 row blocks
    const int bn = blockIdx.x % (C3 / 256);        // 9 col blocks
    const int m0 = bm * 64;
    const int n0 = bn * 256 + wv * 32;
    const int srow = tid >> 2, schunk = (tid & 3) * 8;   // staging: 16B/thread
    const unsigned short* gsrc = xb + (size_t)(m0 + srow) * CDIM + schunk;
    unsigned short* l0 = &smem[0][srow * LDS_STRIDE + schunk];
    unsigned short* l1 = &smem[1][srow * LDS_STRIDE + schunk];

    async_stage16(gsrc, l0);                       // stage k-tile 0 -> buf0

    bf16x16 b0[2], b1[2];
    #pragma unroll
    for (int j = 0; j < 2; ++j)
        b0[j] = load_b_frag(wq + (size_t)(n0 + j * 16) * CDIM, CDIM, lane);

    wait_async0();
    __syncthreads();

    f32x8 acc[4][2] = {};
    for (int kk = 0; kk < CDIM / 32; kk += 2) {
        // ---- even step: consume buf0/b0, stage kk+1 -> buf1/b1 ----
        async_stage16(gsrc + (kk + 1) * 32, l1);
        #pragma unroll
        for (int j = 0; j < 2; ++j)
            b1[j] = load_b_frag(wq + (size_t)(n0 + j * 16) * CDIM + (kk + 1) * 32,
                                CDIM, lane);
        #pragma unroll
        for (int i = 0; i < 4; ++i) {
            bf16x16 a = load_a_frag_lds(&smem[0][i * 16 * LDS_STRIDE], lane);
            #pragma unroll
            for (int j = 0; j < 2; ++j)
                acc[i][j] = wmma_bf16(a, b0[j], acc[i][j]);
        }
        wait_async0();
        __syncthreads();
        // ---- odd step: consume buf1/b1, stage kk+2 -> buf0/b0 ----
        if (kk + 2 < CDIM / 32) {
            async_stage16(gsrc + (kk + 2) * 32, l0);
            #pragma unroll
            for (int j = 0; j < 2; ++j)
                b0[j] = load_b_frag(wq + (size_t)(n0 + j * 16) * CDIM + (kk + 2) * 32,
                                    CDIM, lane);
        }
        #pragma unroll
        for (int i = 0; i < 4; ++i) {
            bf16x16 a = load_a_frag_lds(&smem[1][i * 16 * LDS_STRIDE], lane);
            #pragma unroll
            for (int j = 0; j < 2; ++j)
                acc[i][j] = wmma_bf16(a, b1[j], acc[i][j]);
        }
        wait_async0();
        __syncthreads();
    }

    const float qscale = 0.015625f;                // 4096^-0.5
    const int hl = lane >> 4, nc = lane & 15;
    #pragma unroll
    for (int i = 0; i < 4; ++i) {
        #pragma unroll
        for (int j = 0; j < 2; ++j) {
            const int o = n0 + j * 16 + nc;
            const int which = o / CDIM;            // 0=q 1=k 2=v (wave-uniform)
            const int rem = o % CDIM;
            const int h = rem / HDIM, d = rem % HDIM;
            const float bv = bias[o];
            #pragma unroll
            for (int r = 0; r < 8; ++r) {
                const int t = m0 + i * 16 + hl * 8 + r;
                const int bi = t >> 12, ntok = t & (NTOK - 1);
                const float val = acc[i][j][r] + bv;
                if (which == 0)
                    q_t[(((size_t)(bi * NHEAD + h) * HDIM + d) << 12) + ntok] =
                        f32_to_bf16(val * qscale);
                else if (which == 1)
                    k_t[(((size_t)(bi * NHEAD + h) * HDIM + d) << 12) + ntok] =
                        f32_to_bf16(val);
                else
                    v_b[((size_t)(bi * NHEAD + h) * NTOK + ntok) * HDIM + d] =
                        f32_to_bf16(val);
            }
        }
    }
}

// ---------------------------------------------------------------------------
// attn[b,h] = qT[96 x 4096] @ k[4096 x 96] -> f32 [96 x 96].
// 4 waves per (b,h), each a 48x48 quadrant (3x3 WMMA tiles), K=4096.
// Register double-buffered (2-stage pipeline over 64-wide K steps).
// ---------------------------------------------------------------------------
__global__ __launch_bounds__(128) void attn_gemm_kernel(
        const unsigned short* __restrict__ q_t,
        const unsigned short* __restrict__ k_t,
        float* __restrict__ attn_f) {
    const int lane = threadIdx.x & 31;
    const int wv = threadIdx.x >> 5;               // 0..3
    const int bh = blockIdx.x;                     // 0..127
    const int m0 = (wv >> 1) * 48, n0 = (wv & 1) * 48;
    const unsigned short* A  = q_t + (size_t)bh * HDIM * NTOK;
    const unsigned short* Bt = k_t + (size_t)bh * HDIM * NTOK;

    f32x8 acc[3][3] = {};
    bf16x16 a0[3], b0[3], a1[3], b1[3];
    #pragma unroll
    for (int i = 0; i < 3; ++i)
        a0[i] = load_a_frag(A + (size_t)(m0 + i * 16) * NTOK, NTOK, lane);
    #pragma unroll
    for (int j = 0; j < 3; ++j)
        b0[j] = load_b_frag(Bt + (size_t)(n0 + j * 16) * NTOK, NTOK, lane);

    for (int k0 = 0; k0 < NTOK; k0 += 64) {
        #pragma unroll
        for (int i = 0; i < 3; ++i)
            a1[i] = load_a_frag(A + (size_t)(m0 + i * 16) * NTOK + k0 + 32,
                                NTOK, lane);
        #pragma unroll
        for (int j = 0; j < 3; ++j)
            b1[j] = load_b_frag(Bt + (size_t)(n0 + j * 16) * NTOK + k0 + 32,
                                NTOK, lane);
        #pragma unroll
        for (int i = 0; i < 3; ++i)
            #pragma unroll
            for (int j = 0; j < 3; ++j)
                acc[i][j] = wmma_bf16(a0[i], b0[j], acc[i][j]);
        if (k0 + 64 < NTOK) {
            #pragma unroll
            for (int i = 0; i < 3; ++i)
                a0[i] = load_a_frag(A + (size_t)(m0 + i * 16) * NTOK + k0 + 64,
                                    NTOK, lane);
            #pragma unroll
            for (int j = 0; j < 3; ++j)
                b0[j] = load_b_frag(Bt + (size_t)(n0 + j * 16) * NTOK + k0 + 64,
                                    NTOK, lane);
        }
        #pragma unroll
        for (int i = 0; i < 3; ++i)
            #pragma unroll
            for (int j = 0; j < 3; ++j)
                acc[i][j] = wmma_bf16(a1[i], b1[j], acc[i][j]);
    }

    const int hl = lane >> 4, nc = lane & 15;
    float* out = attn_f + (size_t)bh * HDIM * HDIM;
    #pragma unroll
    for (int i = 0; i < 3; ++i)
        #pragma unroll
        for (int j = 0; j < 3; ++j)
            #pragma unroll
            for (int r = 0; r < 8; ++r)
                out[(size_t)(m0 + i * 16 + hl * 8 + r) * HDIM + (n0 + j * 16 + nc)] =
                    acc[i][j][r];
}

// ---------------------------------------------------------------------------
// Row softmax over attn [B*H*96 rows x 96], emit bf16. One wave per row.
// ---------------------------------------------------------------------------
__global__ __launch_bounds__(256) void softmax_kernel(const float* __restrict__ attn_f,
                                                      unsigned short* __restrict__ attn_b) {
    const int row = blockIdx.x * (blockDim.x >> 5) + (threadIdx.x >> 5);
    const int lane = threadIdx.x & 31;
    const float* p = attn_f + (size_t)row * HDIM;
    float v0 = p[lane], v1 = p[lane + 32], v2 = p[lane + 64];
    float m = fmaxf(v0, fmaxf(v1, v2));
    #pragma unroll
    for (int off = 16; off; off >>= 1) m = fmaxf(m, __shfl_xor(m, off, 32));
    float e0 = __expf(v0 - m), e1 = __expf(v1 - m), e2 = __expf(v2 - m);
    float s = e0 + e1 + e2;
    #pragma unroll
    for (int off = 16; off; off >>= 1) s += __shfl_xor(s, off, 32);
    const float inv = 1.0f / s;
    unsigned short* q = attn_b + (size_t)row * HDIM;
    q[lane]      = f32_to_bf16(e0 * inv);
    q[lane + 32] = f32_to_bf16(e1 * inv);
    q[lane + 64] = f32_to_bf16(e2 * inv);
}

// ---------------------------------------------------------------------------
// out[b,h] = v[4096 x 96] @ attn^T[96 x 96], written into [B, N, C] bf16.
// Wave macro-tile 64x48 (4x3 WMMA tiles), 128 waves per (b,h). K=96 only.
// ---------------------------------------------------------------------------
__global__ __launch_bounds__(256) void av_gemm_kernel(
        const unsigned short* __restrict__ v_b,
        const unsigned short* __restrict__ attn_b,
        unsigned short* __restrict__ out_attn) {
    const int lane = threadIdx.x & 31;
    const int wave = blockIdx.x * (blockDim.x >> 5) + (threadIdx.x >> 5);
    const int bh = wave >> 7;                      // 0..127
    const int ww = wave & 127;
    const int m0 = (ww >> 1) * 64;                 // token tile
    const int n0 = (ww & 1) * 48;                  // d tile
    const unsigned short* A  = v_b    + (size_t)bh * NTOK * HDIM;
    const unsigned short* Bt = attn_b + (size_t)bh * HDIM * HDIM;

    f32x8 acc[4][3] = {};
    #pragma unroll
    for (int kk = 0; kk < 3; ++kk) {
        const int k0 = kk * 32;
        bf16x16 a[4], b[3];
        #pragma unroll
        for (int i = 0; i < 4; ++i)
            a[i] = load_a_frag(A + (size_t)(m0 + i * 16) * HDIM + k0, HDIM, lane);
        #pragma unroll
        for (int j = 0; j < 3; ++j)
            b[j] = load_b_frag(Bt + (size_t)(n0 + j * 16) * HDIM + k0, HDIM, lane);
        #pragma unroll
        for (int i = 0; i < 4; ++i)
            #pragma unroll
            for (int j = 0; j < 3; ++j)
                acc[i][j] = wmma_bf16(a[i], b[j], acc[i][j]);
    }

    const int bi = bh >> 3, h = bh & 7;
    const int hl = lane >> 4, nc = lane & 15;
    #pragma unroll
    for (int i = 0; i < 4; ++i)
        #pragma unroll
        for (int j = 0; j < 3; ++j)
            #pragma unroll
            for (int r = 0; r < 8; ++r) {
                const int ntok = m0 + i * 16 + hl * 8 + r;
                const int d = n0 + j * 16 + nc;
                out_attn[((size_t)bi * NTOK + ntok) * CDIM + h * HDIM + d] =
                    f32_to_bf16(acc[i][j][r]);
            }
}

// ---------------------------------------------------------------------------
// Output projection: [65536 x 768] @ [768 x 768] + bias -> f32 d_out.
// Same LDS-async-staged 2x-unrolled structure as qkv_gemm.
// ---------------------------------------------------------------------------
__global__ __launch_bounds__(256) void proj_gemm_kernel(
        const unsigned short* __restrict__ out_attn,
        const unsigned short* __restrict__ wp,     // [768 x 768] (== B^T)
        const float* __restrict__ bias,
        float* __restrict__ y) {
    __shared__ __align__(16) unsigned short smem[2][64 * LDS_STRIDE];
    const int tid  = threadIdx.x;
    const int lane = tid & 31, wv = tid >> 5;
    const int bm = blockIdx.x / (CDIM / 256);      // 1024 row blocks
    const int bn = blockIdx.x % (CDIM / 256);      // 3 col blocks
    const int m0 = bm * 64;
    const int n0 = bn * 256 + wv * 32;
    const int srow = tid >> 2, schunk = (tid & 3) * 8;
    const unsigned short* gsrc = out_attn + (size_t)(m0 + srow) * CDIM + schunk;
    unsigned short* l0 = &smem[0][srow * LDS_STRIDE + schunk];
    unsigned short* l1 = &smem[1][srow * LDS_STRIDE + schunk];

    async_stage16(gsrc, l0);

    bf16x16 b0[2], b1[2];
    #pragma unroll
    for (int j = 0; j < 2; ++j)
        b0[j] = load_b_frag(wp + (size_t)(n0 + j * 16) * CDIM, CDIM, lane);

    wait_async0();
    __syncthreads();

    f32x8 acc[4][2] = {};
    for (int kk = 0; kk < CDIM / 32; kk += 2) {
        async_stage16(gsrc + (kk + 1) * 32, l1);
        #pragma unroll
        for (int j = 0; j < 2; ++j)
            b1[j] = load_b_frag(wp + (size_t)(n0 + j * 16) * CDIM + (kk + 1) * 32,
                                CDIM, lane);
        #pragma unroll
        for (int i = 0; i < 4; ++i) {
            bf16x16 a = load_a_frag_lds(&smem[0][i * 16 * LDS_STRIDE], lane);
            #pragma unroll
            for (int j = 0; j < 2; ++j)
                acc[i][j] = wmma_bf16(a, b0[j], acc[i][j]);
        }
        wait_async0();
        __syncthreads();
        if (kk + 2 < CDIM / 32) {
            async_stage16(gsrc + (kk + 2) * 32, l0);
            #pragma unroll
            for (int j = 0; j < 2; ++j)
                b0[j] = load_b_frag(wp + (size_t)(n0 + j * 16) * CDIM + (kk + 2) * 32,
                                    CDIM, lane);
        }
        #pragma unroll
        for (int i = 0; i < 4; ++i) {
            bf16x16 a = load_a_frag_lds(&smem[1][i * 16 * LDS_STRIDE], lane);
            #pragma unroll
            for (int j = 0; j < 2; ++j)
                acc[i][j] = wmma_bf16(a, b1[j], acc[i][j]);
        }
        wait_async0();
        __syncthreads();
    }

    const int hl = lane >> 4, nc = lane & 15;
    #pragma unroll
    for (int i = 0; i < 4; ++i)
        #pragma unroll
        for (int j = 0; j < 2; ++j) {
            const int c = n0 + j * 16 + nc;
            const float bv = bias[c];
            #pragma unroll
            for (int r = 0; r < 8; ++r)
                y[(size_t)(m0 + i * 16 + hl * 8 + r) * CDIM + c] = acc[i][j][r] + bv;
        }
}

// ---------------------------------------------------------------------------
extern "C" void kernel_launch(void* const* d_in, const int* in_sizes, int n_in,
                              void* d_out, int out_size, void* d_ws, size_t ws_size,
                              hipStream_t stream) {
    const float* x      = (const float*)d_in[0];
    const float* w_qkv  = (const float*)d_in[1];
    const float* b_qkv  = (const float*)d_in[2];
    const float* w_proj = (const float*)d_in[3];
    const float* b_proj = (const float*)d_in[4];
    float* y = (float*)d_out;

    char* base = (char*)d_ws;
    size_t off = 0;
    auto take = [&](size_t bytes) -> char* {
        char* p = base + off;
        off = (off + bytes + 255) & ~(size_t)255;
        return p;
    };

    unsigned short* xb      = (unsigned short*)take((size_t)TROWS * CDIM * 2);
    unsigned short* wqkvb   = (unsigned short*)take((size_t)C3 * CDIM * 2);
    unsigned short* wprojb  = (unsigned short*)take((size_t)CDIM * CDIM * 2);
    unsigned short* q_t     = (unsigned short*)take((size_t)BATCH * NHEAD * HDIM * NTOK * 2);
    unsigned short* k_t     = (unsigned short*)take((size_t)BATCH * NHEAD * HDIM * NTOK * 2);
    unsigned short* v_b     = (unsigned short*)take((size_t)BATCH * NHEAD * NTOK * HDIM * 2);
    float*          attn_f  = (float*)take((size_t)BATCH * NHEAD * HDIM * HDIM * 4);
    unsigned short* attn_b  = (unsigned short*)take((size_t)BATCH * NHEAD * HDIM * HDIM * 2);
    unsigned short* out_attn = xb;   // xb is dead after qkv_gemm; alias it

    // 1) down-convert inputs to bf16
    cvt_bf16_kernel<<<4096, 256, 0, stream>>>(x, xb, TROWS * CDIM);
    cvt_bf16_kernel<<<1024, 256, 0, stream>>>(w_qkv, wqkvb, C3 * CDIM);
    cvt_bf16_kernel<<<512,  256, 0, stream>>>(w_proj, wprojb, CDIM * CDIM);

    // 2) QKV projection: block tile 64x256, grid 1024*9
    qkv_gemm_kernel<<<9216, 256, 0, stream>>>(xb, wqkvb, b_qkv, q_t, k_t, v_b);

    // 3) channel-attention logits (one block of 4 waves per (b,h))
    attn_gemm_kernel<<<BATCH * NHEAD, 128, 0, stream>>>(q_t, k_t, attn_f);

    // 4) softmax over 12288 rows (8 waves/block)
    softmax_kernel<<<(BATCH * NHEAD * HDIM) / 8, 256, 0, stream>>>(attn_f, attn_b);

    // 5) attn @ v (waves = 128 * 128 = 16384 -> 2048 blocks)
    av_gemm_kernel<<<2048, 256, 0, stream>>>(v_b, attn_b, out_attn);

    // 6) output projection: block tile 64x256, grid 1024*3
    proj_gemm_kernel<<<3072, 256, 0, stream>>>(out_attn, wprojb, b_proj, y);
}
